// RATD_base_49615462203782
// MI455X (gfx1250) — matrix operand order repaired and checked by reading, not compiled, not run on hardware
//
#include <hip/hip_runtime.h>
#include <hip/hip_bf16.h>

// ---------------------------------------------------------------------------
// RATD fused attention for MI455X (gfx1250, wave32, WMMA).
//   B=4 C=16 K=384 L=256 CTX=288 H=8 D=64 INNER=512, BC=64.
// Prologue: convert inputs to f16 and transpose+convert weights (N-major f16)
//           so every WMMA fragment in the hot loops is a contiguous vector
//           load with no v_cvt and no branching in the K-loop.
// Kernel A: per (bc,head) block -> f16 WMMA projections into LDS, then
//           flash-attention with online softmax (f32 accum), f16 output.
//           Projection K-loops are split per concat-segment (x | cond | ref)
//           to eliminate per-iteration segment-select branches.
// Kernel B: output projection (24576x512)@(512x256)+bias via f16 WMMA.
// ---------------------------------------------------------------------------

typedef __attribute__((ext_vector_type(16))) _Float16 v16h;
typedef __attribute__((ext_vector_type(8)))  _Float16 v8h;
typedef __attribute__((ext_vector_type(4)))  _Float16 v4h;
typedef __attribute__((ext_vector_type(8)))  float    v8f;

#define DEV __device__ __forceinline__

static constexpr int KTOK  = 384;
static constexpr int LDIM  = 256;
static constexpr int CTXD  = 288;
static constexpr int HEADS = 8;
static constexpr int DH    = 64;
static constexpr int INNER = 512;
static constexpr int KQ    = 2 * LDIM + CTXD;  // 800
static constexpr int KV    = LDIM + CTXD;      // 544
static constexpr int ROWS  = 64 * KTOK;        // 24576

// ---- workspace layout (offsets in halves, all 32B aligned) ----
static constexpr size_t OXH = 0;                              // attn out f16 (24576x512)
static constexpr size_t OXI = OXH + (size_t)ROWS * INNER;     // x f16   (64*384*256)
static constexpr size_t OCI = OXI + (size_t)64 * KTOK * LDIM; // cond f16
static constexpr size_t ORF = OCI + (size_t)64 * KTOK * LDIM; // ref f16 (4*384*288)
static constexpr size_t OWK = ORF + (size_t)4 * KTOK * CTXD;  // WkT f16 (512x800)
static constexpr size_t OWV = OWK + (size_t)INNER * KQ;       // WvT f16 (512x544)
static constexpr size_t OWO = OWV + (size_t)INNER * KV;       // WoT f16 (256x512)
static constexpr size_t WS_HALVES = OWO + (size_t)LDIM * INNER;

// LDS layout (halves): Q[384*64] | V[384*64] | VT[64*384] | P-scratch[8*512]
static constexpr int SQ_OFF = 0;
static constexpr int SV_OFF = SQ_OFF + KTOK * DH;
static constexpr int VT_OFF = SV_OFF + KTOK * DH;
static constexpr int SP_OFF = VT_OFF + DH * KTOK;
static constexpr int SMEM_BYTES = (SP_OFF + 8 * 16 * 32) * 2;  // 155648 B

DEV v8f wmma_f16(v16h a, v16h b, v8f c) {
  return __builtin_amdgcn_wmma_f32_16x16x32_f16(false, a, false, b,
                                                (short)0, c, false, false);
}

DEV float red_max16(float v) {
  v = fmaxf(v, __shfl_xor(v, 1, 32));
  v = fmaxf(v, __shfl_xor(v, 2, 32));
  v = fmaxf(v, __shfl_xor(v, 4, 32));
  v = fmaxf(v, __shfl_xor(v, 8, 32));
  return v;
}
DEV float red_sum16(float v) {
  v += __shfl_xor(v, 1, 32);
  v += __shfl_xor(v, 2, 32);
  v += __shfl_xor(v, 4, 32);
  v += __shfl_xor(v, 8, 32);
  return v;
}

// 8 contiguous halves at p -> a[base..base+7]
DEV void ld8h(const _Float16* p, v16h& a, int base) {
  v8h t = *(const v8h*)p;
#pragma unroll
  for (int e = 0; e < 8; ++e) a[base + e] = t[e];
}

// one K=32 projection step for 3 m-tiles x 4 n-tiles.
// sp[t]: per-lane row pointer into the current concat segment (local k origin);
// kl: local k offset within segment; WTrow: weight row base for (N0+ln),
// wstride: weight row stride (KQ/KV/INNER); kw: absolute k offset in weights.
DEV void proj_step(const _Float16* s0, const _Float16* s1, const _Float16* s2,
                   int kl, const _Float16* WT, int wstride, int kw,
                   int N0, int ln, int hf, v8f acc[3][4]) {
  v16h a[3];
  ld8h(s0 + kl + 16 * hf, a[0], 0);
  ld8h(s0 + kl + 8 + 16 * hf, a[0], 8);
  ld8h(s1 + kl + 16 * hf, a[1], 0);
  ld8h(s1 + kl + 8 + 16 * hf, a[1], 8);
  ld8h(s2 + kl + 16 * hf, a[2], 0);
  ld8h(s2 + kl + 8 + 16 * hf, a[2], 8);
  __builtin_prefetch(WT + (size_t)(N0 + ln) * wstride + kw + 32, 0, 1);
#pragma unroll
  for (int nt = 0; nt < 4; ++nt) {
    v16h bf = *(const v16h*)(WT + (size_t)(N0 + nt * 16 + ln) * wstride + kw + 16 * hf);
#pragma unroll
    for (int t = 0; t < 3; ++t) acc[t][nt] = wmma_f16(a[t], bf, acc[t][nt]);
  }
}

// ------------------- prologue: convert / transpose ---------------------------
// block ranges: [0,6144) x->f16x4 | [6144,12288) ci | [12288,12720) ref
//               [12720,14320) WkT | [14320,15408) WvT | [15408,15920) WoT
DEV void cvt4(const float* __restrict__ s, _Float16* __restrict__ d, int i) {
  float4 f = ((const float4*)s)[i];
  v4h o;
  o[0] = (_Float16)f.x; o[1] = (_Float16)f.y;
  o[2] = (_Float16)f.z; o[3] = (_Float16)f.w;
  ((v4h*)d)[i] = o;
}
DEV void trcv(const float* __restrict__ w, _Float16* __restrict__ wt,
              int K, int N, int i) {
  int k = i / N, n = i % N;
  wt[(size_t)n * K + k] = (_Float16)w[i];
}

__global__ __launch_bounds__(256) void
ratd_prep(const float* __restrict__ x, const float* __restrict__ ci,
          const float* __restrict__ ref, const float* __restrict__ Wk,
          const float* __restrict__ Wv, const float* __restrict__ Wo,
          _Float16* __restrict__ ws) {
  const int bid = blockIdx.x;
  const int tid = threadIdx.x;
  if (bid < 6144)        cvt4(x,  ws + OXI, bid * 256 + tid);
  else if (bid < 12288)  cvt4(ci, ws + OCI, (bid - 6144) * 256 + tid);
  else if (bid < 12720)  cvt4(ref, ws + ORF, (bid - 12288) * 256 + tid);
  else if (bid < 14320)  trcv(Wk, ws + OWK, KQ, INNER, (bid - 12720) * 256 + tid);
  else if (bid < 15408)  trcv(Wv, ws + OWV, KV, INNER, (bid - 14320) * 256 + tid);
  else                   trcv(Wo, ws + OWO, INNER, LDIM, (bid - 15408) * 256 + tid);
}

// ------------------- kernel A: fused projection + attention ------------------
__global__ __launch_bounds__(256) void
ratd_fused_attn(_Float16* __restrict__ ws) {
  extern __shared__ char smemraw[];
  _Float16* smem = (_Float16*)smemraw;
  _Float16* sQ  = smem + SQ_OFF;
  _Float16* sV  = smem + SV_OFF;
  _Float16* sVT = smem + VT_OFF;

  const int bid  = blockIdx.x;          // 0..511
  const int bc   = bid >> 3;
  const int h    = bid & 7;
  const int b    = bc >> 4;
  const int w    = threadIdx.x >> 5;
  const int lane = threadIdx.x & 31;
  const int ln   = lane & 15;
  const int hf   = lane >> 4;
  const int N0h  = h * DH;

  const _Float16* xh  = ws + OXI + (size_t)bc * (KTOK * LDIM);
  const _Float16* chh = ws + OCI + (size_t)bc * (KTOK * LDIM);
  const _Float16* rh  = ws + ORF + (size_t)b * (KTOK * CTXD);
  const _Float16* WkT = ws + OWK;
  const _Float16* WvT = ws + OWV;
  _Float16* Xh = ws + OXH;

  // per-lane token-row pointers for the wave's 3 m-tiles
  const _Float16 *xrow[3], *crow[3], *rrow[3];
#pragma unroll
  for (int t = 0; t < 3; ++t) {
    const int mrow = (w + 8 * t) * 16 + ln;
    xrow[t] = xh  + (size_t)mrow * LDIM;
    crow[t] = chh + (size_t)mrow * LDIM;
    rrow[t] = rh  + (size_t)mrow * CTXD;
  }

  // ---------------- phase 1: Q projection (segments: x | cond | ref) --------
  {
    v8f qacc[3][4] = {};
    for (int ks = 0; ks < 8; ++ks)          // x segment, abs k [0,256)
      proj_step(xrow[0], xrow[1], xrow[2], ks * 32,
                WkT, KQ, ks * 32, N0h, ln, hf, qacc);
    for (int ks = 0; ks < 8; ++ks)          // cond segment, abs k [256,512)
      proj_step(crow[0], crow[1], crow[2], ks * 32,
                WkT, KQ, 256 + ks * 32, N0h, ln, hf, qacc);
    for (int ks = 0; ks < 9; ++ks)          // ref segment, abs k [512,800)
      proj_step(rrow[0], rrow[1], rrow[2], ks * 32,
                WkT, KQ, 512 + ks * 32, N0h, ln, hf, qacc);
#pragma unroll
    for (int t = 0; t < 3; ++t) {
      const int mt = w + 8 * t;
#pragma unroll
      for (int nt = 0; nt < 4; ++nt)
#pragma unroll
        for (int r = 0; r < 8; ++r)
          sQ[(mt * 16 + r + 8 * hf) * DH + nt * 16 + ln] = (_Float16)qacc[t][nt][r];
    }
  }

  // ---------------- phase 1b: V projection (segments: x | ref) --------------
  {
    v8f vacc[3][4] = {};
    for (int ks = 0; ks < 8; ++ks)          // x segment, abs k [0,256)
      proj_step(xrow[0], xrow[1], xrow[2], ks * 32,
                WvT, KV, ks * 32, N0h, ln, hf, vacc);
    for (int ks = 0; ks < 9; ++ks)          // ref segment, abs k [256,544)
      proj_step(rrow[0], rrow[1], rrow[2], ks * 32,
                WvT, KV, 256 + ks * 32, N0h, ln, hf, vacc);
#pragma unroll
    for (int t = 0; t < 3; ++t) {
      const int mt = w + 8 * t;
#pragma unroll
      for (int nt = 0; nt < 4; ++nt) {
        v8h pk;
#pragma unroll
        for (int r = 0; r < 8; ++r) {
          _Float16 v = (_Float16)vacc[t][nt][r];
          sV[(mt * 16 + r + 8 * hf) * DH + nt * 16 + ln] = v;
          pk[r] = v;
        }
        *(v8h*)(sVT + (nt * 16 + ln) * KTOK + mt * 16 + 8 * hf) = pk;
      }
    }
  }
  __syncthreads();

  // ---------------- phase 2: flash attention --------------------
  const float scale = 0.125f;
  _Float16* sPw = smem + SP_OFF + w * 512;

  for (int t = 0; t < 3; ++t) {
    const int rt = w + 8 * t;
    const _Float16* qrow = sQ + (rt * 16 + ln) * DH;
    v16h qa0, qa1;
    ld8h(qrow + 16 * hf, qa0, 0);
    ld8h(qrow + 8 + 16 * hf, qa0, 8);
    ld8h(qrow + 32 + 16 * hf, qa1, 0);
    ld8h(qrow + 40 + 16 * hf, qa1, 8);

    float mrow[8], lrow[8];
#pragma unroll
    for (int r = 0; r < 8; ++r) { mrow[r] = -__builtin_inff(); lrow[r] = 0.f; }
    v8f o[4] = {};

    for (int jp = 0; jp < KTOK / 32; ++jp) {
      const int j0 = jp * 32;
      v8f s0 = {}, s1 = {};
      {
        const _Float16* vr = sV + (j0 + ln) * DH;
        s0 = wmma_f16(qa0, *(const v16h*)(vr + 16 * hf), s0);
        s0 = wmma_f16(qa1, *(const v16h*)(vr + 32 + 16 * hf), s0);
        const _Float16* vr2 = sV + (j0 + 16 + ln) * DH;
        s1 = wmma_f16(qa0, *(const v16h*)(vr2 + 16 * hf), s1);
        s1 = wmma_f16(qa1, *(const v16h*)(vr2 + 32 + 16 * hf), s1);
      }
      float al[8];
#pragma unroll
      for (int r = 0; r < 8; ++r) {
        s0[r] *= scale; s1[r] *= scale;
        float tm = red_max16(fmaxf(s0[r], s1[r]));
        float mn = fmaxf(mrow[r], tm);
        al[r] = __expf(mrow[r] - mn);
        float p0 = __expf(s0[r] - mn);
        float p1 = __expf(s1[r] - mn);
        s0[r] = p0; s1[r] = p1;
        lrow[r] = lrow[r] * al[r] + red_sum16(p0 + p1);
        mrow[r] = mn;
      }
#pragma unroll
      for (int r = 0; r < 8; ++r) {
        sPw[(r + 8 * hf) * 32 + ln]      = (_Float16)s0[r];
        sPw[(r + 8 * hf) * 32 + 16 + ln] = (_Float16)s1[r];
      }
      asm volatile("s_wait_dscnt 0x0" ::: "memory");
      __builtin_amdgcn_wave_barrier();
      const _Float16* prow = sPw + ln * 32;
      v16h pa;
      ld8h(prow + 16 * hf, pa, 0);
      ld8h(prow + 8 + 16 * hf, pa, 8);
#pragma unroll
      for (int nt = 0; nt < 4; ++nt) {
#pragma unroll
        for (int r = 0; r < 8; ++r) o[nt][r] *= al[r];
        v16h bv = *(const v16h*)(sVT + (nt * 16 + ln) * KTOK + j0 + 16 * hf);
        o[nt] = wmma_f16(pa, bv, o[nt]);
      }
    }
#pragma unroll
    for (int r = 0; r < 8; ++r) lrow[r] = 1.0f / lrow[r];
#pragma unroll
    for (int nt = 0; nt < 4; ++nt)
#pragma unroll
      for (int r = 0; r < 8; ++r)
        Xh[(size_t)(bc * KTOK + rt * 16 + r + 8 * hf) * INNER + N0h + nt * 16 + ln] =
            (_Float16)(o[nt][r] * lrow[r]);
  }
}

// ------------------- kernel B: Y = Xh @ Wo + bo ------------------------------
__global__ __launch_bounds__(256) void
ratd_out_proj(const _Float16* __restrict__ ws, const float* __restrict__ bo,
              float* __restrict__ out) {
  const int bid  = blockIdx.x;           // m-tile 0..1535
  const int w    = threadIdx.x >> 5;
  const int lane = threadIdx.x & 31;
  const int ln   = lane & 15;
  const int hf   = lane >> 4;

  const _Float16* xrow = ws + OXH + (size_t)(bid * 16 + ln) * INNER;
  const _Float16* WoT  = ws + OWO;
  v8f acc[2] = {};
  for (int ks = 0; ks < INNER / 32; ++ks) {
    const int k0 = ks * 32;
    v16h a;
    ld8h(xrow + k0 + 16 * hf, a, 0);
    ld8h(xrow + k0 + 8 + 16 * hf, a, 8);
    if (k0 + 32 < INNER)
      __builtin_prefetch(WoT + (size_t)(w * 32 + ln) * INNER + k0 + 32, 0, 1);
#pragma unroll
    for (int nti = 0; nti < 2; ++nti) {
      const int N0 = (w * 2 + nti) * 16;
      v16h bf = *(const v16h*)(WoT + (size_t)(N0 + ln) * INNER + k0 + 16 * hf);
      acc[nti] = wmma_f16(a, bf, acc[nti]);
    }
  }
#pragma unroll
  for (int nti = 0; nti < 2; ++nti) {
    const int N0 = (w * 2 + nti) * 16;
    const float bias = bo[N0 + ln];
#pragma unroll
    for (int r = 0; r < 8; ++r)
      out[(size_t)(bid * 16 + r + 8 * hf) * LDIM + N0 + ln] = acc[nti][r] + bias;
  }
}

extern "C" void kernel_launch(void* const* d_in, const int* in_sizes, int n_in,
                              void* d_out, int out_size, void* d_ws, size_t ws_size,
                              hipStream_t stream) {
  (void)in_sizes; (void)n_in; (void)out_size; (void)ws_size;
  const float* x   = (const float*)d_in[0];
  const float* ci  = (const float*)d_in[1];
  const float* ref = (const float*)d_in[2];
  const float* Wk  = (const float*)d_in[3];
  const float* Wv  = (const float*)d_in[4];
  const float* Wo  = (const float*)d_in[5];
  const float* bo  = (const float*)d_in[6];
  float* out = (float*)d_out;
  _Float16* ws = (_Float16*)d_ws;   // needs WS_HALVES*2 ~= 53 MB

  ratd_prep<<<15920, 256, 0, stream>>>(x, ci, ref, Wk, Wv, Wo, ws);
  ratd_fused_attn<<<64 * HEADS, 256, SMEM_BYTES, stream>>>(ws);
  ratd_out_proj<<<ROWS / 16, 256, 0, stream>>>(ws, bo, out);
}
